// MPNN_91036126806704
// MI455X (gfx1250) — matrix operand order repaired and checked by reading, not compiled, not run on hardware
//
#include <hip/hip_runtime.h>
#include <hip/hip_bf16.h>
#include <math.h>

// ---------------------------------------------------------------------------
// MPNN forward for MI455X (gfx1250): f16 WMMA GEMMs, async Global->LDS gather,
// L2-resident scatter-mean.
// ---------------------------------------------------------------------------

typedef __attribute__((ext_vector_type(16))) _Float16 v16h;
typedef __attribute__((ext_vector_type(8)))  float    v8f;

#define NN        100000
#define EE        1600000
#define DIM       128
#define WPB       4              // waves per block (wave32)
#define TPB       (WPB * 32)
#define KT_MSG    9              // 288 = 9 * 32  (259 zero-padded)
#define KPAD_MSG  (KT_MSG * 32)
#define KT_UPD    8              // 256 = 8 * 32
#define KPAD_UPD  (KT_UPD * 32)

union FragH { v16h v; uint4 q[2]; };

// Async DMA of one 256 B row (16 x b128) from global into LDS.
// Uses CDNA5 GLOBAL_LOAD_ASYNC_TO_LDS_B128 (ASYNCcnt), GV addressing mode.
__device__ __forceinline__ void async_row_to_lds(unsigned lds_addr,
                                                 unsigned long long gaddr) {
    #pragma unroll
    for (int i = 0; i < 16; ++i) {
        asm volatile("global_load_async_to_lds_b128 %0, %1, off"
                     :: "v"(lds_addr), "v"(gaddr) : "memory");
        lds_addr += 16;
        gaddr    += 16;
    }
}
__device__ __forceinline__ void wait_async() {
    asm volatile("s_wait_asynccnt 0x0" ::: "memory");
}

// --------------------------- weight packing --------------------------------
// Packs W[k][n] (f32, row-major [Ksrc x 128]) into WMMA B-fragment order:
// dst[((kt*8 + nt)*32 + lane)*16 + h] ; per ISA 32x16 f16 B layout:
//   lane 0-15  -> N = lane,    K = kt*32 + (0..7, 16..23)  at halves 0..15
//   lane 16-31 -> N = lane-16, K = kt*32 + (8..15, 24..31) at halves 0..15
__global__ void pack_b_kernel(const float* __restrict__ W,
                              _Float16* __restrict__ dst,
                              int Ksrc, int total) {
    int idx = blockIdx.x * blockDim.x + threadIdx.x;
    if (idx >= total) return;
    int hh   = idx & 15;
    int lane = (idx >> 4) & 31;
    int nt   = (idx >> 9) & 7;
    int kt   = idx >> 12;
    int n = nt * 16 + (lane & 15);
    int k = kt * 32 + ((lane & 16) ? 8 : 0) + (hh & 7) + ((hh & 8) ? 16 : 0);
    float v = (k < Ksrc) ? W[(size_t)k * DIM + n] : 0.0f;
    dst[idx] = (_Float16)v;
}

// ------------------------------ init / zero --------------------------------
__global__ void init_h_kernel(const float* __restrict__ x,
                              float* __restrict__ h,
                              _Float16* __restrict__ hf) {
    int i = blockIdx.x * blockDim.x + threadIdx.x;
    if (i < NN * DIM) { float v = x[i]; h[i] = v; hf[i] = (_Float16)v; }
}

__global__ void zero_kernel(float* __restrict__ agg, float* __restrict__ cnt) {
    int i = blockIdx.x * blockDim.x + threadIdx.x;
    if (i < NN * DIM) agg[i] = 0.0f;
    if (i < NN)       cnt[i] = 0.0f;
}

// --------------------------- edge message kernel ---------------------------
// One wave processes a 16-edge tile:
//   A = [ h[row] | h[col] | (pos[row]-pos[col]) | 0-pad ]  (16 x 288, f16)
//   D = gelu(A @ Wm + bm) -> LayerNorm per row -> atomic scatter into agg/cnt.
__global__ __launch_bounds__(TPB)
void edge_msg_kernel(const _Float16* __restrict__ hf,
                     const float* __restrict__ pos,
                     const long long* __restrict__ erow,
                     const long long* __restrict__ ecol,
                     const _Float16* __restrict__ wmP,   // [9][8][32][16] halves
                     const float* __restrict__ bias,     // [128]
                     float* __restrict__ agg,
                     float* __restrict__ cnt) {
    __shared__ _Float16 lds[WPB][16][KPAD_MSG];          // 4 * 9216 B
    const int  wave = threadIdx.x >> 5;
    const int  lane = threadIdx.x & 31;
    const long tile = (long)blockIdx.x * WPB + wave;
    const long e0   = tile * 16;

    __builtin_prefetch(wmP, 0, 0);                       // global_prefetch_b8

    // ---- stage A tile: 2 lanes per edge-row; gathers go async DMA -> LDS ---
    const int  r     = lane >> 1;
    const int  hsel  = lane & 1;
    const long e     = e0 + r;
    const bool valid = (e < (long)EE);
    {
        if (valid) {
            const long long src = hsel ? ecol[e] : erow[e];
            async_row_to_lds((unsigned)(size_t)&lds[wave][r][hsel * DIM],
                             (unsigned long long)(size_t)(hf + (size_t)src * DIM));
        } else {
            uint4* dp = (uint4*)&lds[wave][r][hsel * DIM];
            uint4 z; z.x = z.y = z.z = z.w = 0u;
            #pragma unroll
            for (int i = 0; i < 16; ++i) dp[i] = z;
        }
        if (hsel == 0) {                                 // tail: edge_attr + pad
            uint4* tz = (uint4*)&lds[wave][r][2 * DIM];
            uint4 z; z.x = z.y = z.z = z.w = 0u;
            #pragma unroll
            for (int i = 0; i < 4; ++i) tz[i] = z;       // zero 32 halves
            if (valid) {
                const long long a = erow[e], b = ecol[e];
                lds[wave][r][2 * DIM + 0] = (_Float16)(pos[a * 3 + 0] - pos[b * 3 + 0]);
                lds[wave][r][2 * DIM + 1] = (_Float16)(pos[a * 3 + 1] - pos[b * 3 + 1]);
                lds[wave][r][2 * DIM + 2] = (_Float16)(pos[a * 3 + 2] - pos[b * 3 + 2]);
            }
        }
    }
    wait_async();                                        // s_wait_asynccnt 0

    // ---- hoist A fragments into VGPRs (ISA 16x32 f16 A layout) ----
    const int am = lane & 15;
    const int g0 = (lane & 16) ? 8 : 0;
    FragH a[KT_MSG];
    const _Float16* arow = &lds[wave][am][0];
    #pragma unroll
    for (int kt = 0; kt < KT_MSG; ++kt) {
        a[kt].q[0] = *(const uint4*)(arow + kt * 32 + g0);
        a[kt].q[1] = *(const uint4*)(arow + kt * 32 + g0 + 16);
    }

    // ---- GEMM (+bias, GELU); reuse staging LDS as the f32 m-tile [16][128] --
    float* mt = (float*)&lds[wave][0][0];
    const int mrow0 = (lane & 16) ? 8 : 0;
    #pragma unroll
    for (int nt = 0; nt < 8; ++nt) {
        const float bn = bias[nt * 16 + am];
        v8f c;
        #pragma unroll
        for (int i = 0; i < 8; ++i) c[i] = bn;
        #pragma unroll
        for (int kt = 0; kt < KT_MSG; ++kt) {
            FragH b;
            const uint4* bp = (const uint4*)(wmP + (((size_t)kt * 8 + nt) * 32 + lane) * 16);
            b.q[0] = bp[0]; b.q[1] = bp[1];
            c = __builtin_amdgcn_wmma_f32_16x16x32_f16(false, a[kt].v, false, b.v,
                                                       (short)0, c, false, false);
        }
        #pragma unroll
        for (int i = 0; i < 8; ++i) {                    // exact-erf GELU
            float xv = c[i];
            float g  = 0.5f * xv * (1.0f + erff(xv * 0.70710678118f));
            mt[(mrow0 + i) * DIM + nt * 16 + am] = g;
        }
    }

    // ---- LayerNorm per edge row + scatter-mean accumulate (L2-resident) ----
    {
        const float* mr = mt + r * DIM + hsel * 64;
        float s = 0.0f, s2 = 0.0f;
        #pragma unroll 8
        for (int i = 0; i < 64; ++i) { float v = mr[i]; s += v; s2 += v * v; }
        s  += __shfl_xor(s,  1, 32);                     // wave32 pair reduce
        s2 += __shfl_xor(s2, 1, 32);
        const float mu   = s * (1.0f / DIM);
        const float var  = s2 * (1.0f / DIM) - mu * mu;
        const float rstd = rsqrtf(var + 1e-5f);
        if (valid) {
            const long long dst = ecol[e];
            float* ap = agg + (size_t)dst * DIM + hsel * 64;
            #pragma unroll 8
            for (int i = 0; i < 64; ++i)
                atomicAdd(&ap[i], (mr[i] - mu) * rstd);
            if (hsel == 0) atomicAdd(&cnt[dst], 1.0f);
        }
    }
}

// ---------------------------- node update kernel ---------------------------
// One wave per 16-node tile: A = [ h | agg/max(cnt,1) ] (16 x 256 f16),
// u = LN(A @ Wu + bu), h += u, refresh hf16 = f16(h).
__global__ __launch_bounds__(TPB)
void node_update_kernel(float* __restrict__ h,
                        _Float16* __restrict__ hf,
                        const float* __restrict__ agg,
                        const float* __restrict__ cnt,
                        const _Float16* __restrict__ wuP,  // [8][8][32][16]
                        const float* __restrict__ bias) {
    __shared__ _Float16 lds[WPB][16][KPAD_UPD];            // 4 * 8192 B
    const int  wave = threadIdx.x >> 5;
    const int  lane = threadIdx.x & 31;
    const long tile = (long)blockIdx.x * WPB + wave;
    const long n0   = tile * 16;

    __builtin_prefetch(wuP, 0, 0);

    const int  r     = lane >> 1;
    const int  hsel  = lane & 1;
    const long node  = n0 + r;
    const bool valid = (node < (long)NN);
    if (valid) {
        if (hsel == 0) {                                   // h row: async DMA
            async_row_to_lds((unsigned)(size_t)&lds[wave][r][0],
                             (unsigned long long)(size_t)(hf + (size_t)node * DIM));
        } else {                                           // agg mean: f32->f16
            const float inv = 1.0f / fmaxf(cnt[node], 1.0f);
            const float* ap = agg + (size_t)node * DIM;
            _Float16* dp = &lds[wave][r][DIM];
            #pragma unroll 8
            for (int i = 0; i < DIM; ++i) dp[i] = (_Float16)(ap[i] * inv);
        }
    } else {
        uint4* dp = (uint4*)&lds[wave][r][hsel * DIM];
        uint4 z; z.x = z.y = z.z = z.w = 0u;
        #pragma unroll
        for (int i = 0; i < 16; ++i) dp[i] = z;
    }
    wait_async();                                          // s_wait_asynccnt 0

    const int am = lane & 15;
    const int g0 = (lane & 16) ? 8 : 0;
    FragH a[KT_UPD];
    const _Float16* arow = &lds[wave][am][0];
    #pragma unroll
    for (int kt = 0; kt < KT_UPD; ++kt) {
        a[kt].q[0] = *(const uint4*)(arow + kt * 32 + g0);
        a[kt].q[1] = *(const uint4*)(arow + kt * 32 + g0 + 16);
    }

    float* mt = (float*)&lds[wave][0][0];
    const int mrow0 = (lane & 16) ? 8 : 0;
    #pragma unroll
    for (int nt = 0; nt < 8; ++nt) {
        const float bn = bias[nt * 16 + am];
        v8f c;
        #pragma unroll
        for (int i = 0; i < 8; ++i) c[i] = bn;
        #pragma unroll
        for (int kt = 0; kt < KT_UPD; ++kt) {
            FragH b;
            const uint4* bp = (const uint4*)(wuP + (((size_t)kt * 8 + nt) * 32 + lane) * 16);
            b.q[0] = bp[0]; b.q[1] = bp[1];
            c = __builtin_amdgcn_wmma_f32_16x16x32_f16(false, a[kt].v, false, b.v,
                                                       (short)0, c, false, false);
        }
        #pragma unroll
        for (int i = 0; i < 8; ++i)
            mt[(mrow0 + i) * DIM + nt * 16 + am] = c[i];
    }

    // LayerNorm + residual + refresh f16 copy
    {
        const float* mr = mt + r * DIM + hsel * 64;
        float s = 0.0f, s2 = 0.0f;
        #pragma unroll 8
        for (int i = 0; i < 64; ++i) { float v = mr[i]; s += v; s2 += v * v; }
        s  += __shfl_xor(s,  1, 32);
        s2 += __shfl_xor(s2, 1, 32);
        const float mu   = s * (1.0f / DIM);
        const float var  = s2 * (1.0f / DIM) - mu * mu;
        const float rstd = rsqrtf(var + 1e-5f);
        if (valid) {
            float*     hp  = h  + (size_t)node * DIM + hsel * 64;
            _Float16*  hfp = hf + (size_t)node * DIM + hsel * 64;
            #pragma unroll 8
            for (int i = 0; i < 64; ++i) {
                float u  = (mr[i] - mu) * rstd;
                float nh = hp[i] + u;
                hp[i]  = nh;
                hfp[i] = (_Float16)nh;
            }
        }
    }
}

// ------------------------------- launcher ----------------------------------
extern "C" void kernel_launch(void* const* d_in, const int* in_sizes, int n_in,
                              void* d_out, int out_size, void* d_ws, size_t ws_size,
                              hipStream_t stream) {
    const float*     x   = (const float*)d_in[0];
    const float*     pos = (const float*)d_in[1];
    const long long* ei  = (const long long*)d_in[2];   // int64 [2][E]
    const float* msg_w[2] = { (const float*)d_in[3], (const float*)d_in[7]  };
    const float* msg_b[2] = { (const float*)d_in[4], (const float*)d_in[8]  };
    const float* upd_w[2] = { (const float*)d_in[5], (const float*)d_in[9]  };
    const float* upd_b[2] = { (const float*)d_in[6], (const float*)d_in[10] };
    float* h = (float*)d_out;                            // h lives in d_out

    // workspace carve (256 B aligned)
    char*  ws  = (char*)d_ws;
    size_t off = 0;
    auto carve = [&](size_t b) -> char* {
        char* p = ws + off;
        off = (off + b + 255) & ~(size_t)255;
        return p;
    };
    _Float16* hf  = (_Float16*)carve((size_t)NN * DIM * sizeof(_Float16));
    float*    agg = (float*)   carve((size_t)NN * DIM * sizeof(float));
    float*    cnt = (float*)   carve((size_t)NN * sizeof(float));
    const int TOTM = KT_MSG * 8 * 32 * 16;               // packed halves (msg)
    const int TOTU = KT_UPD * 8 * 32 * 16;               // packed halves (upd)
    _Float16* wmP[2], * wuP[2];
    wmP[0] = (_Float16*)carve((size_t)TOTM * 2);
    wmP[1] = (_Float16*)carve((size_t)TOTM * 2);
    wuP[0] = (_Float16*)carve((size_t)TOTU * 2);
    wuP[1] = (_Float16*)carve((size_t)TOTU * 2);

    // 1) pack weights into WMMA B-fragment order
    for (int s = 0; s < 2; ++s) {
        pack_b_kernel<<<(TOTM + 255) / 256, 256, 0, stream>>>(msg_w[s], wmP[s], 2 * DIM + 3, TOTM);
        pack_b_kernel<<<(TOTU + 255) / 256, 256, 0, stream>>>(upd_w[s], wuP[s], 2 * DIM,     TOTU);
    }
    // 2) h = x ; hf16 = f16(x)
    init_h_kernel<<<(NN * DIM + 255) / 256, 256, 0, stream>>>(x, h, hf);

    const int ETILES = (EE + 15) / 16;
    const int EBLK   = (ETILES + WPB - 1) / WPB;
    const int NTILES = (NN + 15) / 16;
    const int NBLK   = (NTILES + WPB - 1) / WPB;

    // 3) two message-passing steps
    for (int s = 0; s < 2; ++s) {
        zero_kernel<<<(NN * DIM + 255) / 256, 256, 0, stream>>>(agg, cnt);
        edge_msg_kernel<<<EBLK, TPB, 0, stream>>>(hf, pos, ei, ei + EE,
                                                  wmP[s], msg_b[s], agg, cnt);
        node_update_kernel<<<NBLK, TPB, 0, stream>>>(h, hf, agg, cnt,
                                                     wuP[s], upd_b[s]);
    }
    (void)in_sizes; (void)n_in; (void)out_size; (void)ws_size;
}
